// Sequence_57647051047491
// MI455X (gfx1250) — matrix operand order repaired and checked by reading, not compiled, not run on hardware
//
#include <hip/hip_runtime.h>
#include <hip/hip_bf16.h>

// ---------------------------------------------------------------------------
// 2-layer LSTM (quirk: layer-2 input is c1), B=1024, T=1024, H1=51, H2=1.
// Strategy: 64 blocks x 128 threads (4 waves). Each block owns 16 batch rows.
// Recurrent GEMM h1(16x51) @ Whh1^T(51x204) done with v_wmma_f32_16x16x32_f16:
// K padded to 64 (2 chunks), N padded to 208 (13 tiles split 3/3/3/4 per wave).
// B-operand fragments pre-swizzled by a prologue kernel into d_ws and held
// resident in VGPRs for the whole time loop (wave32 -> large VGPR budget).
// ---------------------------------------------------------------------------

typedef __attribute__((ext_vector_type(16))) _Float16 v16h;
typedef __attribute__((ext_vector_type(8)))  float    v8f;

#define Bsz     1024
#define Tsteps  1024
#define H1n     51
#define NG      204      // 4*H1
#define NTILES  13       // ceil(204/16)
#define CHUNK   64       // x-prefetch chunk (timesteps)

// ---- fast activations ------------------------------------------------------
__device__ __forceinline__ float fsig(float x) {
  x = fminf(fmaxf(x, -30.f), 30.f);
  return __builtin_amdgcn_rcpf(1.f + __expf(-x));
}
__device__ __forceinline__ float ftanh(float x) {
#if __has_builtin(__builtin_amdgcn_tanhf)
  return __builtin_amdgcn_tanhf(x);          // CDNA5 v_tanh_f32
#else
  x = fminf(fmaxf(x, -15.f), 15.f);
  float e = __expf(2.f * x);
  return (e - 1.f) * __builtin_amdgcn_rcpf(e + 1.f);
#endif
}

// ---------------------------------------------------------------------------
// Prologue: convert W_hh1 (204x51 f32, row-major, gates[n] = sum_k h[k]*W[n][k])
// into f16 WMMA B-operand fragments, fragment order [tile][kchunk][lane][elem].
// 16-bit B 32x16 layout: lane%16 = N column; lanes 0-15 hold K {0..7,16..23},
// lanes 16-31 hold K {8..15,24..31} (matches A). kchunk adds +32 to K.
// Total 13*2*32*16 = 13312 halves = 26624 bytes in d_ws.
// ---------------------------------------------------------------------------
__global__ void build_whh1_frags(const float* __restrict__ Whh1,
                                 _Float16* __restrict__ frg) {
  int id = blockIdx.x * 256 + threadIdx.x;
  if (id >= NTILES * 2 * 32 * 16) return;
  int e  = id & 15;
  int l  = (id >> 4) & 31;
  int c  = (id >> 9) & 1;
  int t  = id >> 10;
  int n  = t * 16 + (l & 15);
  int kl = ((e >> 3) << 4) + ((l >> 4) << 3) + (e & 7);
  int k  = c * 32 + kl;
  float v = (n < NG && k < H1n) ? Whh1[n * H1n + k] : 0.f;
  frg[id] = (_Float16)v;
}

// ---------------------------------------------------------------------------
// Main kernel: full time recurrence for 16 batch rows per block.
// ---------------------------------------------------------------------------
__global__ __launch_bounds__(128) void lstm_seq_kernel(
    const float* __restrict__ inp,                     // [B,T]
    const float* __restrict__ Wih1,                    // [204,1]
    const float* __restrict__ bih1,
    const float* __restrict__ bhh1,
    const float* __restrict__ Wih2,                    // [4,51]
    const float* __restrict__ Whh2,                    // [4,1]
    const float* __restrict__ bih2,
    const float* __restrict__ bhh2,
    const _Float16* __restrict__ frg,                  // WMMA B fragments
    float* __restrict__ out)                           // [B,T]
{
  __shared__ _Float16 h1s[16 * 64];        // h1 state, f16, K padded to 64
  __shared__ float    gs[208 * 16];        // gate preacts, [N][M] (M contig)
  __shared__ float    c1s[16 * 52];        // c1 state, row pitch 52
  __shared__ float    xs[16 * CHUNK];      // staged input chunk
  __shared__ float    wih1s[NG], bsum[NG];
  __shared__ float    wih2s[4 * H1n], whh2s[4], b2s[4];
  __shared__ float    g2s[16 * 4];
  __shared__ float    h2s[16], c2s[16];

  const int tid   = threadIdx.x;
  const int lane  = tid & 31;
  const int wave  = tid >> 5;
  const int brow0 = blockIdx.x * 16;

  // ---- init state + stage tiny params into LDS ----
  for (int i = tid; i < 16 * 64; i += 128) h1s[i] = (_Float16)0.f;
  for (int i = tid; i < 16 * 52; i += 128) c1s[i] = 0.f;
  if (tid < 16) { h2s[tid] = 0.f; c2s[tid] = 0.f; }
  for (int i = tid; i < NG; i += 128) {
    wih1s[i] = Wih1[i];
    bsum[i]  = bih1[i] + bhh1[i];
    wih2s[i] = Wih2[i];                    // 4*51 == 204 == NG
  }
  if (tid < 4) { whh2s[tid] = Whh2[tid]; b2s[tid] = bih2[tid] + bhh2[tid]; }

  // ---- resident B fragments (per-wave tile slice: 3/3/3/4) ----
  const int t0 = (wave == 3) ? 9 : wave * 3;
  const int nt = (wave == 3) ? 4 : 3;
  union BF { v16h v; uint4 u[2]; };
  BF bf[4][2];
#pragma unroll
  for (int tt = 0; tt < 4; ++tt)
#pragma unroll
    for (int c = 0; c < 2; ++c) {
      if (tt < nt) {
        const uint4* p =
            (const uint4*)(frg + (size_t)(((t0 + tt) * 2 + c) * 32 + lane) * 16);
        bf[tt][c].u[0] = p[0];
        bf[tt][c].u[1] = p[1];
      } else {
        bf[tt][c].u[0] = make_uint4(0, 0, 0, 0);
        bf[tt][c].u[1] = make_uint4(0, 0, 0, 0);
      }
    }

  // ---- x prefetch registers (chunk 0) ----
  const int xr = tid >> 3;             // row 0..15
  const int xc = (tid & 7) * 8;        // 8 floats per thread
  float4 px0, px1;
  {
    const float4* g = (const float4*)(inp + (size_t)(brow0 + xr) * Tsteps + xc);
    px0 = g[0]; px1 = g[1];
  }

  // ---- A-fragment LDS base (16-bit A 16x32 layout) ----
  // lane%16 = row M; lanes 0-15 take K offsets {0..7,16..23},
  // lanes 16-31 take {8..15,24..31}; chunk1 adds +32 to K.
  const _Float16* abase = h1s + (lane & 15) * 64 + ((lane < 16) ? 0 : 8);
  const int mb = (lane < 16) ? 0 : 8;  // C/D row base for this lane

  __syncthreads();

  for (int tc = 0; tc < Tsteps; tc += CHUNK) {
    // commit prefetched chunk to LDS, then launch next chunk's global loads
    *(float4*)(xs + xr * CHUNK + xc)     = px0;
    *(float4*)(xs + xr * CHUNK + xc + 4) = px1;
    if (tc + CHUNK < Tsteps) {
      const float4* g =
          (const float4*)(inp + (size_t)(brow0 + xr) * Tsteps + (tc + CHUNK) + xc);
      px0 = g[0]; px1 = g[1];
    }
    __syncthreads();

    for (int s = 0; s < CHUNK; ++s) {
      const int t = tc + s;

      // ---- build A fragments from h1 LDS (4x ds_load_b128) ----
      union AF { v16h v; uint4 u[2]; };
      AF a0, a1;
      a0.u[0] = *(const uint4*)(abase);
      a0.u[1] = *(const uint4*)(abase + 16);
      a1.u[0] = *(const uint4*)(abase + 32);
      a1.u[1] = *(const uint4*)(abase + 48);

      // ---- recurrent GEMM: per-wave N tiles, 2 K-chunks each ----
#pragma unroll
      for (int tt = 0; tt < 4; ++tt) {
        if (tt < nt) {
          v8f acc = {};
          acc = __builtin_amdgcn_wmma_f32_16x16x32_f16(
              false, a0.v, false, bf[tt][0].v, (short)0, acc, false, false);
          acc = __builtin_amdgcn_wmma_f32_16x16x32_f16(
              false, a1.v, false, bf[tt][1].v, (short)0, acc, false, false);
          // C layout: this lane = column n, rows mb..mb+7 -> contiguous in gs
          const int n = (t0 + tt) * 16 + (lane & 15);
          float* dst = gs + n * 16 + mb;
          *(float4*)(dst)     = make_float4(acc[0], acc[1], acc[2], acc[3]);
          *(float4*)(dst + 4) = make_float4(acc[4], acc[5], acc[6], acc[7]);
        }
      }
      __syncthreads();

      // ---- layer-1 elementwise: 16 rows x 51 hidden = 816 items ----
#pragma unroll
      for (int k = 0; k < 7; ++k) {
        const int idx = tid + k * 128;
        if (idx < 16 * H1n) {
          const int m = idx & 15;
          const int j = idx >> 4;
          const float x  = xs[m * CHUNK + s];
          const float pi = gs[(j)        * 16 + m] + x * wih1s[j]        + bsum[j];
          const float pf = gs[(H1n + j)  * 16 + m] + x * wih1s[H1n + j]  + bsum[H1n + j];
          const float pg = gs[(2*H1n + j)* 16 + m] + x * wih1s[2*H1n + j]+ bsum[2*H1n + j];
          const float po = gs[(3*H1n + j)* 16 + m] + x * wih1s[3*H1n + j]+ bsum[3*H1n + j];
          const float cn = fsig(pf) * c1s[m * 52 + j] + fsig(pi) * ftanh(pg);
          c1s[m * 52 + j] = cn;
          h1s[m * 64 + j] = (_Float16)(fsig(po) * ftanh(cn));
        }
      }
      __syncthreads();

      // ---- layer-2 gate dots: 16 rows x 4 gates over 51 (input is c1!) ----
      if (tid < 64) {
        const int m = tid >> 2, gate = tid & 3;
        const float* crow = c1s + m * 52;
        const float* wrow = wih2s + gate * H1n;
        float acc2 = 0.f;
#pragma unroll
        for (int j = 0; j < H1n; ++j) acc2 = fmaf(crow[j], wrow[j], acc2);
        g2s[m * 4 + gate] = acc2;
      }
      __syncthreads();

      // ---- layer-2 cell update + output store ----
      if (tid < 16) {
        const int m = tid;
        const float h2 = h2s[m];
        const float pi = g2s[m * 4 + 0] + h2 * whh2s[0] + b2s[0];
        const float pf = g2s[m * 4 + 1] + h2 * whh2s[1] + b2s[1];
        const float pg = g2s[m * 4 + 2] + h2 * whh2s[2] + b2s[2];
        const float po = g2s[m * 4 + 3] + h2 * whh2s[3] + b2s[3];
        const float c2n = fsig(pf) * c2s[m] + fsig(pi) * ftanh(pg);
        c2s[m] = c2n;
        h2s[m] = fsig(po) * ftanh(c2n);
        out[(size_t)(brow0 + m) * Tsteps + t] = c2n;   // fire-and-forget
      }
      __syncthreads();
    }
  }
}

// ---------------------------------------------------------------------------
extern "C" void kernel_launch(void* const* d_in, const int* in_sizes, int n_in,
                              void* d_out, int out_size, void* d_ws, size_t ws_size,
                              hipStream_t stream) {
  const float* inp  = (const float*)d_in[0];
  const float* Wih1 = (const float*)d_in[1];
  const float* Whh1 = (const float*)d_in[2];
  const float* bih1 = (const float*)d_in[3];
  const float* bhh1 = (const float*)d_in[4];
  const float* Wih2 = (const float*)d_in[5];
  const float* Whh2 = (const float*)d_in[6];
  const float* bih2 = (const float*)d_in[7];
  const float* bhh2 = (const float*)d_in[8];
  // d_in[9] = future (0) — ignored.

  _Float16* frg = (_Float16*)d_ws;                 // 26,624 bytes used

  // Prologue: swizzle Whh1 into WMMA B-fragment order (f16).
  build_whh1_frags<<<52, 256, 0, stream>>>(Whh1, frg);

  // Main: 64 blocks of 16 batch rows, 4 waves each.
  lstm_seq_kernel<<<Bsz / 16, 128, 0, stream>>>(
      inp, Wih1, bih1, bhh1, Wih2, Whh2, bih2, bhh2, frg, (float*)d_out);
}
